// LlamaAttention_1632087572781
// MI455X (gfx1250) — matrix operand order repaired and checked by reading, not compiled, not run on hardware
//
#include <hip/hip_runtime.h>

typedef __attribute__((ext_vector_type(16))) __bf16 v16bf;
typedef __attribute__((ext_vector_type(2)))  __bf16 v2bf;
typedef __attribute__((ext_vector_type(8)))  float  v8f;
typedef __attribute__((ext_vector_type(2)))  float  v2f;
typedef __attribute__((ext_vector_type(8)))  unsigned short v8us;
typedef __attribute__((ext_vector_type(4)))  float  v4f;
typedef __attribute__((ext_vector_type(4)))  unsigned int v4u;
typedef __attribute__((ext_vector_type(8)))  int v8i;
typedef __attribute__((ext_vector_type(4)))  int v4i;

namespace {
constexpr int kBz = 2, kS = 2048, kD = 2048, kH = 16, kHD = 128;
constexpr size_t kTensor = (size_t)kBz * kH * kS * kHD;  // 8388608 elements
}

// Native bf16 conversions (lower to v_cvt_pk_bf16_f32 / scalar cvt on gfx1250)
__device__ __forceinline__ unsigned short f2bf(float f) {
  return __builtin_bit_cast(unsigned short, (__bf16)f);
}
__device__ __forceinline__ unsigned pk2bf(float x, float y) {
  v2bf v; v[0] = (__bf16)x; v[1] = (__bf16)y;
  return __builtin_bit_cast(unsigned, v);
}

union FragBF { v16bf v; v8us h[2]; };

__device__ __forceinline__ v8f wmma_bf16(const FragBF& a, const FragBF& b, v8f c) {
  return __builtin_amdgcn_wmma_f32_16x16x32_bf16(false, a.v, false, b.v, (short)0, c,
                                                 false, false);
}

// ---------------------------------------------------------------------------
// Kernel 1: QKV projection. C[n,m] = sum_k X[n,k] * W[m,k]  (NT GEMM)
// X: 4096x2048 f32 row-major, W: 2048x2048 f32 row-major.
// Output written fp32 in (B,H,S,HD) layout. blockIdx.z selects q/k/v.
// Block tile: 128(M) x 64(N), K-step 32, 4 waves (2x2), wave tile 64x32.
// ---------------------------------------------------------------------------
__global__ __launch_bounds__(128) void qkv_proj_kernel(
    const float* __restrict__ Xq, const float* __restrict__ Xk,
    const float* __restrict__ Xv, const float* __restrict__ Wq,
    const float* __restrict__ Wk, const float* __restrict__ Wv,
    float* __restrict__ Oq, float* __restrict__ Ok, float* __restrict__ Ov) {
  const float* X; const float* W; float* O;
  if (blockIdx.z == 0)      { X = Xq; W = Wq; O = Oq; }
  else if (blockIdx.z == 1) { X = Xk; W = Wk; O = Ok; }
  else                      { X = Xv; W = Wv; O = Ov; }

  __shared__ __align__(16) unsigned short As[128 * 32];
  __shared__ __align__(16) unsigned short Bs[64 * 32];

  const int tid  = threadIdx.x;
  const int wave = tid >> 5;
  const int lane = tid & 31;
  const int wm = wave >> 1, wn = wave & 1;

  const v8f vzero = {0.f, 0.f, 0.f, 0.f, 0.f, 0.f, 0.f, 0.f};
  v8f acc[4][2];
#pragma unroll
  for (int i = 0; i < 4; ++i)
#pragma unroll
    for (int j = 0; j < 2; ++j) acc[i][j] = vzero;

  const int m0 = blockIdx.x * 128;
  const int n0 = blockIdx.y * 64;
  const int half8 = (lane & 16) ? 8 : 0;     // A-fragment K half offset
  const int halfB = (lane & 16) ? 16 : 0;    // B-fragment K half offset

  for (int k0 = 0; k0 < kD; k0 += 32) {
    __syncthreads();
    // Stage A tile (128 rows x 32 k) fp32 -> bf16
#pragma unroll
    for (int i = 0; i < 8; ++i) {
      int c = tid + i * 128;            // 1024 chunks of 4 floats
      int row = c >> 3, cc = c & 7;
      v4f f = *(const v4f*)(X + (size_t)(m0 + row) * kD + k0 + cc * 4);
      unsigned long long pk =
          (unsigned long long)pk2bf(f.x, f.y) |
          ((unsigned long long)pk2bf(f.z, f.w) << 32);
      *(unsigned long long*)(&As[row * 32 + cc * 4]) = pk;
    }
    // Stage B tile (64 rows of W x 32 k)
#pragma unroll
    for (int i = 0; i < 4; ++i) {
      int c = tid + i * 128;            // 512 chunks
      int row = c >> 3, cc = c & 7;
      v4f f = *(const v4f*)(W + (size_t)(n0 + row) * kD + k0 + cc * 4);
      unsigned long long pk =
          (unsigned long long)pk2bf(f.x, f.y) |
          ((unsigned long long)pk2bf(f.z, f.w) << 32);
      *(unsigned long long*)(&Bs[row * 32 + cc * 4]) = pk;
    }
    __syncthreads();

    FragBF a[4], b[2];
#pragma unroll
    for (int mi = 0; mi < 4; ++mi) {
      int row = wm * 64 + mi * 16 + (lane & 15);
      a[mi].h[0] = *(const v8us*)&As[row * 32 + half8];
      a[mi].h[1] = *(const v8us*)&As[row * 32 + half8 + 16];
    }
#pragma unroll
    for (int ni = 0; ni < 2; ++ni) {
      int col = wn * 32 + ni * 16 + (lane & 15);
      b[ni].h[0] = *(const v8us*)&Bs[col * 32 + half8];
      b[ni].h[1] = *(const v8us*)&Bs[col * 32 + half8 + 16];
    }
#pragma unroll
    for (int mi = 0; mi < 4; ++mi)
#pragma unroll
      for (int ni = 0; ni < 2; ++ni) acc[mi][ni] = wmma_bf16(a[mi], b[ni], acc[mi][ni]);
  }

  // Epilogue: C layout -> (B,H,S,HD) fp32
#pragma unroll
  for (int mi = 0; mi < 4; ++mi)
#pragma unroll
    for (int ni = 0; ni < 2; ++ni)
#pragma unroll
      for (int r = 0; r < 8; ++r) {
        int n = m0 + wm * 64 + mi * 16 + r + ((lane & 16) ? 8 : 0);
        int m = n0 + wn * 32 + ni * 16 + (lane & 15);
        int bb = n >> 11, ss = n & 2047, hh = m >> 7, dd = m & 127;
        O[(((size_t)(bb * kH + hh)) * kS + ss) * kHD + dd] = acc[mi][ni][r];
      }
}

// ---------------------------------------------------------------------------
// Kernel 2: RoPE + bf16 conversion.
// ---------------------------------------------------------------------------
__global__ __launch_bounds__(256) void rope_kernel(
    const float* __restrict__ qf, const float* __restrict__ kf,
    const float* __restrict__ vf, float* __restrict__ kout,
    unsigned short* __restrict__ qbf, unsigned short* __restrict__ kbf,
    unsigned short* __restrict__ vbf) {
  const int NP = (int)(kTensor / 2);
  int idx = blockIdx.x * 256 + threadIdx.x;
  if (idx >= NP) return;
  int d2 = idx & 63;
  int s = (idx >> 6) & 2047;
  size_t base = (size_t)idx * 2;

  float fr = __expf((float)(2 * d2) * (-9.210340371976184f / 128.0f));
  float ang = (float)s * fr;
  float c = __cosf(ang), sn = __sinf(ang);

  const float scale = 0.088388347648318447f;  // 1/sqrt(128)
  v2f q2 = *(const v2f*)(qf + base);
  float qor = q2.x * c + q2.y * sn;
  float qoi = q2.y * c - q2.x * sn;
  *(unsigned*)(qbf + base) = pk2bf(qor * scale, qoi * scale);

  v2f k2 = *(const v2f*)(kf + base);
  float kor = k2.x * c + k2.y * sn;
  float koi = k2.y * c - k2.x * sn;
  v2f ko; ko.x = kor; ko.y = koi;
  *(v2f*)(kout + base) = ko;
  *(unsigned*)(kbf + base) = pk2bf(kor, koi);

  v2f vv = *(const v2f*)(vf + base);
  *(unsigned*)(vbf + base) = pk2bf(vv.x, vv.y);
}

// ---------------------------------------------------------------------------
// Kernel 3: causal flash attention, bf16 WMMA, fp32 online softmax.
// Grid: (S/64, H, B). 128 threads = 4 waves; wave handles 16 q rows.
// K tile staged by the Tensor Data Mover (TDM); V tile staged transposed.
// ---------------------------------------------------------------------------
__global__ __launch_bounds__(128) void flash_attn_kernel(
    const unsigned short* __restrict__ qbf, const unsigned short* __restrict__ kbf,
    const unsigned short* __restrict__ vbf, unsigned short* __restrict__ abf) {
  __shared__ __align__(16) unsigned short Kt[32 * 128];    // [key][d]
  __shared__ __align__(16) unsigned short Vt[128 * 32];    // [d][key]
  __shared__ __align__(16) unsigned short Pb[4 * 16 * 32]; // per-wave P tile

  const int tid = threadIdx.x;
  const int wave = tid >> 5;
  const int lane = tid & 31;
  const int qb = blockIdx.x, h = blockIdx.y, b = blockIdx.z;
  const size_t base = ((size_t)(b * kH + h)) * kS * kHD;
  const int q0 = qb * 64 + wave * 16;

  // Q fragments for the 4 K-steps (d = 0..127), A-matrix layout.
  FragBF qfrag[4];
  {
    int row = q0 + (lane & 15);
    int off8 = (lane & 16) ? 8 : 0;
#pragma unroll
    for (int ks = 0; ks < 4; ++ks) {
      const unsigned short* p = qbf + base + (size_t)row * kHD + ks * 32 + off8;
      qfrag[ks].h[0] = *(const v8us*)p;
      qfrag[ks].h[1] = *(const v8us*)(p + 16);
    }
  }

  const v8f vzero = {0.f, 0.f, 0.f, 0.f, 0.f, 0.f, 0.f, 0.f};
  v8f o[8];
#pragma unroll
  for (int dt = 0; dt < 8; ++dt) o[dt] = vzero;
  float m_i[8], l_i[8];
#pragma unroll
  for (int r = 0; r < 8; ++r) { m_i[r] = -3.0e38f; l_i[r] = 0.f; }

  const int halfB = (lane & 16) ? 16 : 0;
  const int kb_end = 2 * qb + 2;  // 32-key blocks covering causal range

  for (int kb = 0; kb < kb_end; ++kb) {
    __syncthreads();
    // --- Stage K tile (32 keys x 128 d bf16) via Tensor Data Mover ---
    if (wave == 0) {
      unsigned lds_k = (unsigned)(size_t)(void*)Kt;
      unsigned long long ga =
          (unsigned long long)(const void*)(kbf + base + (size_t)(kb * 32) * kHD);
      v4u g0;
      g0[0] = 1u;                                    // count=1, user mode
      g0[1] = lds_k;                                 // lds_addr
      g0[2] = (unsigned)(ga & 0xFFFFFFFFu);          // global_addr[31:0]
      g0[3] = (unsigned)((ga >> 32) & 0x01FFFFFFu)   // global_addr[56:32]
              | (2u << 30);                          // type=2 ("image")
      v8i g1;
      g1[0] = (int)(1u << 16);      // data_size = 1 -> 2 bytes (bf16)
      g1[1] = (int)(128u << 16);    // tensor_dim0 = 128 (low 16 at bit 48)
      g1[2] = (int)(2048u << 16);   // tensor_dim1 = 2048 (low 16 at bit 80)
      g1[3] = (int)(128u << 16);    // tile_dim0 = 128 (bits 127:112)
      g1[4] = 32;                   // tile_dim1 = 32
      g1[5] = 128;                  // tensor_dim0_stride = 128 (low 32)
      g1[6] = 0;
      g1[7] = 0;
      v4i gz4 = {0, 0, 0, 0};
      v8i gz8 = {0, 0, 0, 0, 0, 0, 0, 0};
      __builtin_amdgcn_tensor_load_to_lds(g0, g1, gz4, gz4, gz8, 0);
      __builtin_amdgcn_s_wait_tensorcnt((short)0);
    }
    // --- Stage V tile transposed: Vt[d][key] (b128 loads, u16 scatter) ---
#pragma unroll
    for (int i = 0; i < 4; ++i) {
      int c = tid + i * 128;               // 512 chunks of 8 bf16
      int key = c >> 4, coff = (c & 15) * 8;
      v8us vv = *(const v8us*)(vbf + base + (size_t)(kb * 32 + key) * kHD + coff);
#pragma unroll
      for (int j = 0; j < 8; ++j) Vt[(coff + j) * 32 + key] = vv[j];
    }
    __syncthreads();

    // S = Q K^T for two 16-key tiles.
    v8f s0 = vzero, s1 = vzero;
#pragma unroll
    for (int ks = 0; ks < 4; ++ks) {
      FragBF b0, b1;
      int c0 = (lane & 15), doff = ks * 32 + (halfB >> 1);
      b0.h[0] = *(const v8us*)&Kt[c0 * 128 + doff];
      b0.h[1] = *(const v8us*)&Kt[c0 * 128 + doff + 16];
      b1.h[0] = *(const v8us*)&Kt[(16 + c0) * 128 + doff];
      b1.h[1] = *(const v8us*)&Kt[(16 + c0) * 128 + doff + 16];
      s0 = wmma_bf16(qfrag[ks], b0, s0);
      s1 = wmma_bf16(qfrag[ks], b1, s1);
    }

    // Causal mask + online softmax. VGPR r holds row (q0 + r + 8*(lane>=16)),
    // lane&15 is the key column within the tile.
    const int rowbase = q0 + ((lane & 16) ? 8 : 0);
    const int keybase = kb * 32 + (lane & 15);
    unsigned short* P = &Pb[wave * 512];
#pragma unroll
    for (int r = 0; r < 8; ++r) {
      float a0 = s0[r], a1 = s1[r];
      int row = rowbase + r;
      if (keybase > row)      a0 = -1.0e30f;
      if (keybase + 16 > row) a1 = -1.0e30f;
      float mx = fmaxf(a0, a1);
      mx = fmaxf(mx, __shfl_xor(mx, 1, 32));
      mx = fmaxf(mx, __shfl_xor(mx, 2, 32));
      mx = fmaxf(mx, __shfl_xor(mx, 4, 32));
      mx = fmaxf(mx, __shfl_xor(mx, 8, 32));
      float mn = fmaxf(m_i[r], mx);
      float corr = __expf(m_i[r] - mn);
      float p0 = __expf(a0 - mn), p1 = __expf(a1 - mn);
      float rs = p0 + p1;
      rs += __shfl_xor(rs, 1, 32);
      rs += __shfl_xor(rs, 2, 32);
      rs += __shfl_xor(rs, 4, 32);
      rs += __shfl_xor(rs, 8, 32);
      l_i[r] = l_i[r] * corr + rs;
      m_i[r] = mn;
#pragma unroll
      for (int dt = 0; dt < 8; ++dt) o[dt][r] *= corr;
      int prow = r + ((lane & 16) ? 8 : 0);
      P[prow * 32 + (lane & 15)]      = f2bf(p0);
      P[prow * 32 + 16 + (lane & 15)] = f2bf(p1);
    }

    // P fragment (A layout) from per-wave LDS region (same-wave, DScnt-ordered).
    FragBF pf;
    {
      int row = lane & 15;
      int off8 = (lane & 16) ? 8 : 0;
      pf.h[0] = *(const v8us*)&P[row * 32 + off8];
      pf.h[1] = *(const v8us*)&P[row * 32 + off8 + 16];
    }
    // O += P V  (8 d-tiles)
#pragma unroll
    for (int dt = 0; dt < 8; ++dt) {
      FragBF vf;
      int d = dt * 16 + (lane & 15);
      vf.h[0] = *(const v8us*)&Vt[d * 32 + (halfB >> 1)];
      vf.h[1] = *(const v8us*)&Vt[d * 32 + (halfB >> 1) + 16];
      o[dt] = wmma_bf16(pf, vf, o[dt]);
    }
  }

  // Epilogue: normalize and write bf16 into (B,S,D) attention buffer.
  float rinv[8];
#pragma unroll
  for (int r = 0; r < 8; ++r) rinv[r] = __builtin_amdgcn_rcpf(l_i[r]);
#pragma unroll
  for (int dt = 0; dt < 8; ++dt)
#pragma unroll
    for (int r = 0; r < 8; ++r) {
      int srow = q0 + r + ((lane & 16) ? 8 : 0);
      int col = h * kHD + dt * 16 + (lane & 15);
      abf[((size_t)(b * kS + srow)) * kD + col] = f2bf(o[dt][r] * rinv[r]);
    }
}

// ---------------------------------------------------------------------------
// Kernel 4: output projection. out[n,m] = sum_k A_bf16[n,k] * Wo[m,k], fp32 out.
// ---------------------------------------------------------------------------
__global__ __launch_bounds__(128) void out_proj_kernel(
    const unsigned short* __restrict__ A, const float* __restrict__ W,
    float* __restrict__ C) {
  __shared__ __align__(16) unsigned short As[128 * 32];
  __shared__ __align__(16) unsigned short Bs[64 * 32];

  const int tid = threadIdx.x;
  const int wave = tid >> 5;
  const int lane = tid & 31;
  const int wm = wave >> 1, wn = wave & 1;

  const v8f vzero = {0.f, 0.f, 0.f, 0.f, 0.f, 0.f, 0.f, 0.f};
  v8f acc[4][2];
#pragma unroll
  for (int i = 0; i < 4; ++i)
#pragma unroll
    for (int j = 0; j < 2; ++j) acc[i][j] = vzero;

  const int m0 = blockIdx.x * 128;
  const int n0 = blockIdx.y * 64;
  const int half8 = (lane & 16) ? 8 : 0;
  const int halfB = (lane & 16) ? 16 : 0;

  for (int k0 = 0; k0 < kD; k0 += 32) {
    __syncthreads();
#pragma unroll
    for (int i = 0; i < 4; ++i) {
      int c = tid + i * 128;               // 512 chunks of 8 bf16
      int row = c >> 2, cc = c & 3;
      *(v8us*)&As[row * 32 + cc * 8] =
          *(const v8us*)(A + (size_t)(m0 + row) * kD + k0 + cc * 8);
    }
#pragma unroll
    for (int i = 0; i < 4; ++i) {
      int c = tid + i * 128;
      int row = c >> 3, cc = c & 7;
      v4f f = *(const v4f*)(W + (size_t)(n0 + row) * kD + k0 + cc * 4);
      unsigned long long pk =
          (unsigned long long)pk2bf(f.x, f.y) |
          ((unsigned long long)pk2bf(f.z, f.w) << 32);
      *(unsigned long long*)(&Bs[row * 32 + cc * 4]) = pk;
    }
    __syncthreads();

    FragBF a[4], b[2];
#pragma unroll
    for (int mi = 0; mi < 4; ++mi) {
      int row = wm * 64 + mi * 16 + (lane & 15);
      a[mi].h[0] = *(const v8us*)&As[row * 32 + half8];
      a[mi].h[1] = *(const v8us*)&As[row * 32 + half8 + 16];
    }
#pragma unroll
    for (int ni = 0; ni < 2; ++ni) {
      int col = wn * 32 + ni * 16 + (lane & 15);
      b[ni].h[0] = *(const v8us*)&Bs[col * 32 + half8];
      b[ni].h[1] = *(const v8us*)&Bs[col * 32 + half8 + 16];
    }
#pragma unroll
    for (int mi = 0; mi < 4; ++mi)
#pragma unroll
      for (int ni = 0; ni < 2; ++ni) acc[mi][ni] = wmma_bf16(a[mi], b[ni], acc[mi][ni]);
  }

#pragma unroll
  for (int mi = 0; mi < 4; ++mi)
#pragma unroll
    for (int ni = 0; ni < 2; ++ni)
#pragma unroll
      for (int r = 0; r < 8; ++r) {
        int n = m0 + wm * 64 + mi * 16 + r + ((lane & 16) ? 8 : 0);
        int m = n0 + wn * 32 + ni * 16 + (lane & 15);
        C[(size_t)n * kD + m] = acc[mi][ni][r];
      }
}

// ---------------------------------------------------------------------------
extern "C" void kernel_launch(void* const* d_in, const int* in_sizes, int n_in,
                              void* d_out, int out_size, void* d_ws, size_t ws_size,
                              hipStream_t stream) {
  (void)in_sizes; (void)n_in; (void)out_size; (void)ws_size;
  const float* queries = (const float*)d_in[0];
  const float* keys    = (const float*)d_in[1];
  const float* values  = (const float*)d_in[2];
  const float* Wq      = (const float*)d_in[3];
  const float* Wk      = (const float*)d_in[4];
  const float* Wv      = (const float*)d_in[5];
  const float* Wo      = (const float*)d_in[6];
  // d_in[7] = mask (unused; causal mask computed analytically)

  float* out  = (float*)d_out;              // (B,S,D)
  float* kout = out + kTensor;              // (B,H,S,HD) post-RoPE
  float* vout = kout + kTensor;             // (B,H,S,HD)

  // Workspace layout
  float* qf32 = (float*)d_ws;               // (B,H,S,HD) pre-RoPE
  float* kf32 = qf32 + kTensor;
  unsigned short* qbf = (unsigned short*)(kf32 + kTensor);
  unsigned short* kbf = qbf + kTensor;
  unsigned short* vbf = kbf + kTensor;
  unsigned short* abf = vbf + kTensor;      // attention output bf16 (B,S,D)

  // 1) QKV projections (z = 0:q, 1:k, 2:v)
  qkv_proj_kernel<<<dim3(32, 32, 3), 128, 0, stream>>>(
      queries, keys, values, Wq, Wk, Wv, qf32, kf32, vout);

  // 2) RoPE + bf16 conversions
  {
    int np = (int)(kTensor / 2);
    rope_kernel<<<(np + 255) / 256, 256, 0, stream>>>(qf32, kf32, vout, kout,
                                                      qbf, kbf, vbf);
  }

  // 3) Causal flash attention
  flash_attn_kernel<<<dim3(kS / 64, kH, kBz), 128, 0, stream>>>(qbf, kbf, vbf, abf);

  // 4) Output projection
  out_proj_kernel<<<dim3(32, 32, 1), 128, 0, stream>>>(abf, Wo, out);
}
